// PointNet_FP_Module_2430951489689
// MI455X (gfx1250) — matrix operand order repaired and checked by reading, not compiled, not run on hardware
//
#include <hip/hip_runtime.h>

// ---------------------------------------------------------------------------
// Problem dimensions (fixed by reference setup_inputs)
// ---------------------------------------------------------------------------
#define B_   8
#define N1_  8192
#define N2_  2048
#define C1_  128          // points1 channels
#define C2_  256          // points2 channels
#define CIN_ 384          // C1_+C2_ (GEMM0 K)
#define CMID_ 256         // GEMM0 output channels
#define COUT_ 128         // GEMM1 output channels
#define M_   (B_*N1_)     // 65536 rows

typedef __attribute__((ext_vector_type(16))) __bf16 v16bf;
typedef __attribute__((ext_vector_type(8)))  float  v8f;
typedef __attribute__((ext_vector_type(4)))  int    v4i_;

#define AS1_ __attribute__((address_space(1)))
#define AS3_ __attribute__((address_space(3)))

// Async global->LDS path (gfx1250): guarded so both host pass and any
// toolchain lacking the builtins still compile (fallback = VGPR copy loop).
#if defined(__has_builtin)
#if __has_builtin(__builtin_amdgcn_global_load_async_to_lds_b128) && \
    __has_builtin(__builtin_amdgcn_s_wait_asynccnt)
#define HAVE_ASYNC_LDS 1
#endif
#endif

// ---------------------------------------------------------------------------
// helpers
// ---------------------------------------------------------------------------
__device__ __forceinline__ unsigned short f2bf(float f) {
    union { float f; unsigned int u; } v; v.f = f;
    unsigned int u = v.u;
    unsigned int r = u + 0x7FFFu + ((u >> 16) & 1u);   // round-to-nearest-even
    return (unsigned short)(r >> 16);
}
__device__ __forceinline__ unsigned int pack2bf(float a, float b) {
    return (unsigned int)f2bf(a) | ((unsigned int)f2bf(b) << 16);
}

// ---------------------------------------------------------------------------
// K0: fp32 -> bf16 conversion (weights)
// ---------------------------------------------------------------------------
__global__ void cvt_f32_bf16(const float* __restrict__ src,
                             unsigned short* __restrict__ dst, int n) {
    int i = blockIdx.x * blockDim.x + threadIdx.x;
    if (i < n) dst[i] = f2bf(src[i]);
}

// ---------------------------------------------------------------------------
// K1: brute-force 3-NN (xyz2 slice in LDS) + inverse-distance interpolation
//     + concat [interp(256) | points1(128)] written as bf16 A-matrix.
//     grid = B_*32 blocks of 256 threads; block's b-slice shares LDS xyz2.
//     LDS fill uses GLOBAL_LOAD_ASYNC_TO_LDS_B128 + s_wait_asynccnt when
//     available (CDNA5 async path, no VGPR round-trip).
// ---------------------------------------------------------------------------
__global__ __launch_bounds__(256) void knn_interp_concat(
    const float* __restrict__ xyz1, const float* __restrict__ xyz2,
    const float* __restrict__ points1, const float* __restrict__ points2,
    unsigned short* __restrict__ Xb)   // (M_, CIN_) bf16 row-major
{
    __shared__ float s_xyz[N2_ * 3];   // 24 KB of the 320 KB WGP LDS

    const int b = blockIdx.x >> 5;                       // 32 blocks per batch
    const int n = ((blockIdx.x & 31) << 8) + threadIdx.x;

#if defined(HAVE_ASYNC_LDS)
    {
        // 6144 floats = 1536 x 16B; 6 async b128 issues per lane.
        char* gsrc = (char*)(xyz2 + (size_t)b * N2_ * 3);
        #pragma unroll
        for (int t = 0; t < 6; ++t) {
            const int idx = threadIdx.x + t * 256;       // 16B-chunk index
            __builtin_amdgcn_global_load_async_to_lds_b128(
                (AS1_ v4i_*)(gsrc + (size_t)idx * 16),
                (AS3_ v4i_*)((char*)s_xyz + (size_t)idx * 16),
                0, 0);
        }
        __builtin_amdgcn_s_wait_asynccnt(0);             // drain this wave's ASYNCcnt
    }
#else
    for (int i = threadIdx.x; i < N2_ * 3; i += 256)
        s_xyz[i] = xyz2[(size_t)b * N2_ * 3 + i];
#endif
    __syncthreads();                                     // all waves' fills visible

    const size_t row = (size_t)b * N1_ + n;
    const float px = xyz1[row * 3 + 0];
    const float py = xyz1[row * 3 + 1];
    const float pz = xyz1[row * 3 + 2];

    float d0 = 3.0e38f, d1 = 3.0e38f, d2 = 3.0e38f;
    int   i0 = 0, i1 = 0, i2 = 0;
    #pragma unroll 4
    for (int j = 0; j < N2_; ++j) {
        const float dx = px - s_xyz[j * 3 + 0];
        const float dy = py - s_xyz[j * 3 + 1];
        const float dz = pz - s_xyz[j * 3 + 2];
        const float d  = dx * dx + dy * dy + dz * dz;
        if (d < d2) {
            if (d < d1) {
                if (d < d0) { d2 = d1; i2 = i1; d1 = d0; i1 = i0; d0 = d; i0 = j; }
                else        { d2 = d1; i2 = i1; d1 = d;  i1 = j; }
            } else          { d2 = d;  i2 = j; }
        }
    }

    const float r0 = 1.0f / (d0 + 1e-8f);
    const float r1 = 1.0f / (d1 + 1e-8f);
    const float r2 = 1.0f / (d2 + 1e-8f);
    const float rs = 1.0f / (r0 + r1 + r2);
    const float w0 = r0 * rs, w1 = r1 * rs, w2 = r2 * rs;

    const float* g0p = points2 + ((size_t)b * N2_ + i0) * C2_;
    const float* g1p = points2 + ((size_t)b * N2_ + i1) * C2_;
    const float* g2p = points2 + ((size_t)b * N2_ + i2) * C2_;

    unsigned int* xr = (unsigned int*)(Xb + row * CIN_);   // 4B-aligned
    #pragma unroll 4
    for (int c = 0; c < C2_; c += 2) {
        const float a = w0 * g0p[c]     + w1 * g1p[c]     + w2 * g2p[c];
        const float d = w0 * g0p[c + 1] + w1 * g1p[c + 1] + w2 * g2p[c + 1];
        xr[c >> 1] = pack2bf(a, d);
    }
    const float* s1 = points1 + row * C1_;
    #pragma unroll 4
    for (int c = 0; c < C1_; c += 2)
        xr[(C2_ + c) >> 1] = pack2bf(s1[c], s1[c + 1]);
}

// ---------------------------------------------------------------------------
// K2/K6: bf16 WMMA GEMM.  C(M x Cout) = A(M x K, bf16) * W(Cout x K, bf16)^T
// Each wave computes a 16(M) x 64(N) tile: 4 fp32 accumulators, K-step 32.
// Fragment layouts per CDNA5 ISA 7.12.2:
//   A 16x32 bf16: lane half 0 -> K {0..7,16..23}, half 1 -> K {8..15,24..31},
//                 row M = lane&15.
//   B 32x16 bf16: lane holds column N = lane&15; half 0 -> K {0..7,16..23},
//                 half 1 -> K {8..15,24..31} (same order as A).
//   C/D 16x16 f32: VGPR r -> M = (lane>>4)*8 + r, N = lane&15.
// ---------------------------------------------------------------------------
union Frag { uint4 u[2]; v16bf v; };

__global__ __launch_bounds__(256) void gemm_wmma_bf16(
    const unsigned short* __restrict__ A,
    const unsigned short* __restrict__ W,
    float* __restrict__ C,
    int M, int K, int Cout)
{
    const int wave = threadIdx.x >> 5;
    const int lane = threadIdx.x & 31;
    const int half = lane >> 4;
    const int lrow = lane & 15;

    const int strips = Cout >> 6;                          // 64-wide N strips
    const int tid    = blockIdx.x * 8 + wave;              // wave-tile id
    const int mtile  = tid / strips;
    const int nbase  = (tid % strips) << 6;

    const unsigned short* arow = A + (size_t)(mtile * 16 + lrow) * K;

    v8f z = {0.f, 0.f, 0.f, 0.f, 0.f, 0.f, 0.f, 0.f};
    v8f acc0 = z, acc1 = z, acc2 = z, acc3 = z;

    for (int k0 = 0; k0 < K; k0 += 32) {
        Frag a;
        a.u[0] = *(const uint4*)(arow + k0 + half * 8);        // K k0+{0..7}/{8..15}
        a.u[1] = *(const uint4*)(arow + k0 + 16 + half * 8);   // K k0+{16..23}/{24..31}

        #pragma unroll
        for (int s = 0; s < 4; ++s) {
            const unsigned short* brow =
                W + (size_t)(nbase + s * 16 + lrow) * K;
            Frag bf;
            bf.u[0] = *(const uint4*)(brow + k0 + half * 8);
            bf.u[1] = *(const uint4*)(brow + k0 + 16 + half * 8);
            v8f& acc = (s == 0) ? acc0 : (s == 1) ? acc1 : (s == 2) ? acc2 : acc3;
            acc = __builtin_amdgcn_wmma_f32_16x16x32_bf16(
                false, a.v, false, bf.v, (short)0, acc, false, false);
        }
    }

    #pragma unroll
    for (int s = 0; s < 4; ++s) {
        const v8f acc = (s == 0) ? acc0 : (s == 1) ? acc1 : (s == 2) ? acc2 : acc3;
        float* cp = C + (size_t)(mtile * 16 + half * 8) * Cout + nbase + s * 16 + lrow;
        #pragma unroll
        for (int r = 0; r < 8; ++r) cp[(size_t)r * Cout] = acc[r];
    }
}

// ---------------------------------------------------------------------------
// K3/K7: per-block column partial sums (deterministic BN stats, stage 1).
// grid = R blocks, block = Cc threads; block reduces rows [blk*rows_per, ...).
// ---------------------------------------------------------------------------
__global__ void col_partial(const float* __restrict__ H,
                            float* __restrict__ psum, float* __restrict__ psq,
                            int Cc, int rows_per)
{
    const int c  = threadIdx.x;
    const int r0 = blockIdx.x * rows_per;
    float s = 0.f, q = 0.f;
    for (int r = 0; r < rows_per; ++r) {
        const float v = H[(size_t)(r0 + r) * Cc + c];
        s += v; q += v * v;
    }
    psum[blockIdx.x * Cc + c] = s;
    psq [blockIdx.x * Cc + c] = q;
}

// K4/K8: finalize BN scale/shift per channel (stage 2).
__global__ void bn_finalize(const float* __restrict__ psum,
                            const float* __restrict__ psq,
                            const float* __restrict__ g, const float* __restrict__ b,
                            float* __restrict__ scale, float* __restrict__ shift,
                            int Cc, int R, float invM)
{
    const int c = threadIdx.x;
    if (c >= Cc) return;
    float s = 0.f, q = 0.f;
    for (int r = 0; r < R; ++r) { s += psum[r * Cc + c]; q += psq[r * Cc + c]; }
    const float mean = s * invM;
    const float var  = q * invM - mean * mean;          // biased, matches jnp.var
    const float sc   = g[c] * rsqrtf(var + 1e-5f);
    scale[c] = sc;
    shift[c] = b[c] - mean * sc;
}

// K5: BN+ReLU -> bf16 A-matrix for the next GEMM (two channels per thread).
// Cc must be a power of two (256 here).
__global__ void bn_relu_bf16(const float* __restrict__ H,
                             const float* __restrict__ scale,
                             const float* __restrict__ shift,
                             unsigned short* __restrict__ Xo,
                             int Cc, long total)
{
    const long i = (long)blockIdx.x * blockDim.x + threadIdx.x;
    const long e = i * 2;
    if (e >= total) return;
    const int c = (int)e & (Cc - 1);
    const float v0 = fmaxf(H[e]     * scale[c]     + shift[c],     0.f);
    const float v1 = fmaxf(H[e + 1] * scale[c + 1] + shift[c + 1], 0.f);
    ((unsigned int*)Xo)[i] = pack2bf(v0, v1);
}

// K9: BN+ReLU -> fp32 output.  Cc must be a power of two (128 here).
__global__ void bn_relu_f32(const float* __restrict__ H,
                            const float* __restrict__ scale,
                            const float* __restrict__ shift,
                            float* __restrict__ out,
                            int Cc, long total)
{
    const long e = (long)blockIdx.x * blockDim.x + threadIdx.x;
    if (e >= total) return;
    const int c = (int)e & (Cc - 1);
    out[e] = fmaxf(H[e] * scale[c] + shift[c], 0.f);
}

// ---------------------------------------------------------------------------
// launch
// ---------------------------------------------------------------------------
extern "C" void kernel_launch(void* const* d_in, const int* in_sizes, int n_in,
                              void* d_out, int out_size, void* d_ws, size_t ws_size,
                              hipStream_t stream)
{
    const float* xyz1    = (const float*)d_in[0];
    const float* xyz2    = (const float*)d_in[1];
    const float* points1 = (const float*)d_in[2];
    const float* points2 = (const float*)d_in[3];
    const float* W0      = (const float*)d_in[4];
    const float* g0      = (const float*)d_in[5];
    const float* b0      = (const float*)d_in[6];
    const float* W1      = (const float*)d_in[7];
    const float* g1      = (const float*)d_in[8];
    const float* b1      = (const float*)d_in[9];
    float* out = (float*)d_out;

    // ---- workspace layout (bytes) ----
    char* ws = (char*)d_ws;
    size_t off = 0;
    auto take = [&](size_t bytes) { size_t o = off; off = (off + bytes + 255) & ~(size_t)255; return o; };
    unsigned short* Xb  = (unsigned short*)(ws + take((size_t)M_ * CIN_ * 2));   // 50.3 MB
    unsigned short* W0b = (unsigned short*)(ws + take((size_t)CMID_ * CIN_ * 2));
    unsigned short* W1b = (unsigned short*)(ws + take((size_t)COUT_ * CMID_ * 2));
    float* H0   = (float*)(ws + take((size_t)M_ * CMID_ * 4));                   // 67.1 MB
    unsigned short* X1b = (unsigned short*)(ws + take((size_t)M_ * CMID_ * 2));  // 33.5 MB
    float* H1   = (float*)(ws + take((size_t)M_ * COUT_ * 4));                   // 33.5 MB
    const int R = 256;                                                            // partial blocks
    float* P0s  = (float*)(ws + take((size_t)R * CMID_ * 4));
    float* P0q  = (float*)(ws + take((size_t)R * CMID_ * 4));
    float* P1s  = (float*)(ws + take((size_t)R * COUT_ * 4));
    float* P1q  = (float*)(ws + take((size_t)R * COUT_ * 4));
    float* Sc0  = (float*)(ws + take(CMID_ * 4));
    float* Sh0  = (float*)(ws + take(CMID_ * 4));
    float* Sc1  = (float*)(ws + take(COUT_ * 4));
    float* Sh1  = (float*)(ws + take(COUT_ * 4));
    (void)ws_size; (void)in_sizes; (void)n_in; (void)out_size;

    // K0: weights -> bf16
    cvt_f32_bf16<<<(CMID_ * CIN_ + 255) / 256, 256, 0, stream>>>(W0, W0b, CMID_ * CIN_);
    cvt_f32_bf16<<<(COUT_ * CMID_ + 255) / 256, 256, 0, stream>>>(W1, W1b, COUT_ * CMID_);

    // K1: 3-NN + interpolate + concat (bf16 A-matrix)
    knn_interp_concat<<<B_ * (N1_ / 256), 256, 0, stream>>>(xyz1, xyz2, points1, points2, Xb);

    // K2: GEMM0  (65536 x 384) x (384 x 256)
    {
        const int waves = (M_ / 16) * (CMID_ / 64);    // 16384
        gemm_wmma_bf16<<<waves / 8, 256, 0, stream>>>(Xb, W0b, H0, M_, CIN_, CMID_);
    }
    // K3/K4: BN0 stats
    col_partial<<<R, CMID_, 0, stream>>>(H0, P0s, P0q, CMID_, M_ / R);
    bn_finalize<<<1, CMID_, 0, stream>>>(P0s, P0q, g0, b0, Sc0, Sh0, CMID_, R, 1.0f / (float)M_);
    // K5: BN0+ReLU -> bf16
    {
        const long total = (long)M_ * CMID_;
        bn_relu_bf16<<<(unsigned)((total / 2 + 255) / 256), 256, 0, stream>>>(H0, Sc0, Sh0, X1b, CMID_, total);
    }
    // K6: GEMM1  (65536 x 256) x (256 x 128)
    {
        const int waves = (M_ / 16) * (COUT_ / 64);    // 8192
        gemm_wmma_bf16<<<waves / 8, 256, 0, stream>>>(X1b, W1b, H1, M_, CMID_, COUT_);
    }
    // K7/K8: BN1 stats
    col_partial<<<R, COUT_, 0, stream>>>(H1, P1s, P1q, COUT_, M_ / R);
    bn_finalize<<<1, COUT_, 0, stream>>>(P1s, P1q, g1, b1, Sc1, Sh1, COUT_, R, 1.0f / (float)M_);
    // K9: BN1+ReLU -> fp32 output
    {
        const long total = (long)M_ * COUT_;
        bn_relu_f32<<<(unsigned)((total + 255) / 256), 256, 0, stream>>>(H1, Sc1, Sh1, out, COUT_, total);
    }
}
